// MoELayer_48816598286567
// MI455X (gfx1250) — matrix operand order repaired and checked by reading, not compiled, not run on hardware
//
#include <hip/hip_runtime.h>
#include <math.h>

// ---------------------------------------------------------------------------
// MoE forward for MI455X (gfx1250, wave32, WMMA).
// Heavy GEMMs run on v_wmma_f32_16x16x32_bf16 (f32 accum), outer-product
// register tiling: each wave holds B fragments resident and reuses them across
// 4 M-subtiles (block tile 64M x 128N) to cut L2 weight traffic 4x.
// fp32 weights/activations are converted to bf16 once per call (~0.75 GB
// streaming, ~33us at 23.3 TB/s) -- the GEMMs themselves are compute-bound
// (~210 GFLOP, ~400 FLOP/byte of HBM traffic), so the bf16 matrix pipe
// (8x the K-depth of the only f32 WMMA shape) is the right target.
// ---------------------------------------------------------------------------

#define Hh   2560
#define Ee   8
#define Ii   1664
#define I2   3328   // 2*I (gate+up)
#define ISs  3328   // shared intermediate
#define Ss   2048   // tokens
#define MT   4      // M-subtiles per wave (64 rows per block tile)

typedef __attribute__((ext_vector_type(16))) __bf16 v16bf;
typedef __attribute__((ext_vector_type(8)))  __bf16 v8bf;
typedef __attribute__((ext_vector_type(8)))  float  v8f;

union AFrag { v16bf v; v8bf h[2]; };

// A fragment (16x32 bf16, row-major source, per-lane row pointer).
// Lane l: m = l&15, half = l>>4. VGPRs0-3 hold K = 16*half+0..7,
// VGPRs4-7 hold K = 16*half+8..15  -> two contiguous 16B loads.
__device__ __forceinline__ v16bf load_a_frag(const __bf16* row, int k0, int half) {
  AFrag f;
  f.h[0] = *(const v8bf*)(row + k0 + 16 * half);
  f.h[1] = *(const v8bf*)(row + k0 + 16 * half + 8);
  return f.v;
}

// B fragment (32x16 bf16) from W stored row-major [N][K].
// Lane l: n = l&15, half = l>>4; holds K = 16*half + 0..15 of row n
// -> one contiguous 32B load.
__device__ __forceinline__ v16bf load_b_frag(const __bf16* wrow, int k0, int half) {
  return *(const v16bf*)(wrow + k0 + 16 * half);
}

__device__ __forceinline__ v8f wmma_bf16(v16bf a, v16bf b, v8f c) {
  return __builtin_amdgcn_wmma_f32_16x16x32_bf16(false, a, false, b, (short)0, c,
                                                 false, false);
}

__device__ __forceinline__ float silu(float g) {
  return g / (1.0f + __expf(-g));
}

// ---------------------------------------------------------------- conversion
__global__ void cvt_f32_bf16(const float* __restrict__ src,
                             __bf16* __restrict__ dst, int n) {
  int i = (blockIdx.x * blockDim.x + threadIdx.x) * 4;
  if (i >= n) return;
  float4 v = *(const float4*)(src + i);
  union { __bf16 b[4]; uint2 u; } o;
  o.b[0] = (__bf16)v.x; o.b[1] = (__bf16)v.y;
  o.b[2] = (__bf16)v.z; o.b[3] = (__bf16)v.w;
  *(uint2*)(dst + i) = o.u;
}

// ---------------------------------------------------------------- router
// One block per token: logits over 8 experts, softmax, top-2, normalized w.
__global__ void router_kernel(const float* __restrict__ x,
                              const float* __restrict__ rw,
                              float* __restrict__ topw, int* __restrict__ topi) {
  int t = blockIdx.x, tid = threadIdx.x;
  float p[Ee];
#pragma unroll
  for (int e = 0; e < Ee; ++e) p[e] = 0.0f;
  for (int hh = tid; hh < Hh; hh += 256) {
    float xv = x[(size_t)t * Hh + hh];
#pragma unroll
    for (int e = 0; e < Ee; ++e) p[e] += xv * rw[e * Hh + hh];
  }
  __shared__ float sm[Ee * 256];
#pragma unroll
  for (int e = 0; e < Ee; ++e) sm[e * 256 + tid] = p[e];
  __syncthreads();
  for (int st = 128; st > 0; st >>= 1) {
    if (tid < st)
#pragma unroll
      for (int e = 0; e < Ee; ++e) sm[e * 256 + tid] += sm[e * 256 + tid + st];
    __syncthreads();
  }
  if (tid == 0) {
    float lg[Ee], mx = -1e30f;
#pragma unroll
    for (int e = 0; e < Ee; ++e) { lg[e] = sm[e * 256]; mx = fmaxf(mx, lg[e]); }
    float se = 0.0f, pr[Ee];
#pragma unroll
    for (int e = 0; e < Ee; ++e) { pr[e] = __expf(lg[e] - mx); se += pr[e]; }
    int i0 = 0;
    for (int e = 1; e < Ee; ++e) if (pr[e] > pr[i0]) i0 = e;
    int i1 = (i0 == 0) ? 1 : 0;
    for (int e = 0; e < Ee; ++e) if (e != i0 && pr[e] > pr[i1]) i1 = e;
    float w0 = pr[i0] / se, w1 = pr[i1] / se, sw = w0 + w1;
    topw[t * 2 + 0] = w0 / sw; topw[t * 2 + 1] = w1 / sw;
    topi[t * 2 + 0] = i0;      topi[t * 2 + 1] = i1;
  }
}

// ------------------------------------------------------ deterministic lists
// One wave; thread e scans tokens in order -> stable per-expert lists.
__global__ void build_lists(const int* __restrict__ topi, int* __restrict__ tok,
                            int* __restrict__ slot, int* __restrict__ cnt,
                            int* __restrict__ base) {
  __shared__ int scnt[Ee];
  int e = threadIdx.x;
  if (e < Ee) {
    int c = 0;
    for (int t = 0; t < Ss; ++t) {
      if (topi[2 * t] == e)      { tok[e * Ss + c] = t; slot[e * Ss + c] = 0; ++c; }
      else if (topi[2 * t + 1] == e) { tok[e * Ss + c] = t; slot[e * Ss + c] = 1; ++c; }
    }
    scnt[e] = c;
  }
  __syncthreads();
  if (threadIdx.x == 0) {
    int acc = 0;
    for (int i = 0; i < Ee; ++i) { cnt[i] = scnt[i]; base[i] = acc; acc += scnt[i]; }
  }
}

// -------------------------------------------- expert gate/up GEMM + SiLU*up
// grid: (I/128, Ss/64, E), block 256 (8 waves, each a 16N column strip).
// Wave keeps Bg/Bu fragments resident and sweeps MT=4 M-subtiles (64 rows),
// then applies SiLU(g)*u and writes bf16 activations packed by expert.
__global__ void gemm_gateup_expert(const __bf16* __restrict__ xb,
                                   const __bf16* __restrict__ w1b,
                                   const int* __restrict__ tok,
                                   const int* __restrict__ cnt,
                                   const int* __restrict__ base,
                                   __bf16* __restrict__ actb) {
  int e = blockIdx.z;
  int ce = cnt[e];
  int m0 = blockIdx.y * (16 * MT);
  if (m0 >= ce) return;                       // block-uniform: EXEC stays full
  int wv = threadIdx.x >> 5, l = threadIdx.x & 31;
  int half = l >> 4, q = l & 15;
  int nb = blockIdx.x * 128 + wv * 16;        // i-space column base
  const __bf16* arow[MT];
#pragma unroll
  for (int i = 0; i < MT; ++i) {
    int row = m0 + 16 * i + q;
    if (row > ce - 1) row = ce - 1;
    arow[i] = xb + (size_t)tok[e * Ss + row] * Hh;
  }
  const __bf16* bg = w1b + ((size_t)e * I2 + (nb + q)) * Hh;       // gate rows
  const __bf16* bu = bg + (size_t)Ii * Hh;                          // up rows
  v8f cg[MT] = {}, cu[MT] = {};
  for (int k0 = 0; k0 < Hh; k0 += 32) {
    v16bf Bg = load_b_frag(bg, k0, half);
    v16bf Bu = load_b_frag(bu, k0, half);
#pragma unroll
    for (int i = 0; i < MT; ++i) {
      v16bf a = load_a_frag(arow[i], k0, half);
      cg[i] = wmma_bf16(a, Bg, cg[i]);
      cu[i] = wmma_bf16(a, Bu, cu[i]);
    }
  }
  int be = base[e];
#pragma unroll
  for (int i = 0; i < MT; ++i)
#pragma unroll
    for (int j = 0; j < 8; ++j) {
      int r = m0 + 16 * i + j + 8 * half;     // C layout: M = j + 8*half
      if (r < ce) {
        float v = silu(cg[i][j]) * cu[i][j];
        actb[(size_t)(be + r) * Ii + nb + q] = (__bf16)v;
      }
    }
}

// ------------------------------------------------------ expert down GEMM
// grid: (H/128, Ss/64, E). Writes unscaled f32 rows to ybuf[token][slot][H].
__global__ void gemm_down_expert(const __bf16* __restrict__ actb,
                                 const __bf16* __restrict__ w2b,
                                 const int* __restrict__ tok,
                                 const int* __restrict__ slot,
                                 const int* __restrict__ cnt,
                                 const int* __restrict__ base,
                                 float* __restrict__ ybuf) {
  int e = blockIdx.z;
  int ce = cnt[e];
  int m0 = blockIdx.y * (16 * MT);
  if (m0 >= ce) return;
  int wv = threadIdx.x >> 5, l = threadIdx.x & 31;
  int half = l >> 4, q = l & 15;
  int nb = blockIdx.x * 128 + wv * 16;        // H column base
  int be = base[e];
  const __bf16* arow[MT];
#pragma unroll
  for (int i = 0; i < MT; ++i) {
    int row = m0 + 16 * i + q;
    if (row > ce - 1) row = ce - 1;
    arow[i] = actb + (size_t)(be + row) * Ii;
  }
  const __bf16* brow = w2b + ((size_t)e * Hh + (nb + q)) * Ii;
  v8f c[MT] = {};
  for (int k0 = 0; k0 < Ii; k0 += 32) {
    v16bf b = load_b_frag(brow, k0, half);
#pragma unroll
    for (int i = 0; i < MT; ++i) {
      v16bf a = load_a_frag(arow[i], k0, half);
      c[i] = wmma_bf16(a, b, c[i]);
    }
  }
#pragma unroll
  for (int i = 0; i < MT; ++i)
#pragma unroll
    for (int j = 0; j < 8; ++j) {
      int r = m0 + 16 * i + j + 8 * half;
      if (r < ce) {
        int t = tok[e * Ss + r], s = slot[e * Ss + r];
        ybuf[((size_t)t * 2 + s) * Hh + nb + q] = c[i][j];
      }
    }
}

// ---------------------------------------------- shared expert gate/up GEMM
// grid: (IS/128, Ss/64).
__global__ void gemm_shared_gateup(const __bf16* __restrict__ xb,
                                   const __bf16* __restrict__ sgb,
                                   const __bf16* __restrict__ sub,
                                   __bf16* __restrict__ acts) {
  int m0 = blockIdx.y * (16 * MT);
  int wv = threadIdx.x >> 5, l = threadIdx.x & 31;
  int half = l >> 4, q = l & 15;
  int nb = blockIdx.x * 128 + wv * 16;
  const __bf16* bg = sgb + (size_t)(nb + q) * Hh;
  const __bf16* bu = sub + (size_t)(nb + q) * Hh;
  v8f cg[MT] = {}, cu[MT] = {};
  for (int k0 = 0; k0 < Hh; k0 += 32) {
    v16bf Bg = load_b_frag(bg, k0, half);
    v16bf Bu = load_b_frag(bu, k0, half);
#pragma unroll
    for (int i = 0; i < MT; ++i) {
      v16bf a = load_a_frag(xb + (size_t)(m0 + 16 * i + q) * Hh, k0, half);
      cg[i] = wmma_bf16(a, Bg, cg[i]);
      cu[i] = wmma_bf16(a, Bu, cu[i]);
    }
  }
#pragma unroll
  for (int i = 0; i < MT; ++i)
#pragma unroll
    for (int j = 0; j < 8; ++j) {
      int r = m0 + 16 * i + j + 8 * half;
      acts[(size_t)r * ISs + nb + q] = (__bf16)(silu(cg[i][j]) * cu[i][j]);
    }
}

// ------------------------------------------------ shared expert down GEMM
// grid: (H/128, Ss/64). Writes shared_out straight into d_out (initializes it).
__global__ void gemm_shared_down(const __bf16* __restrict__ acts,
                                 const __bf16* __restrict__ sdb,
                                 float* __restrict__ out) {
  int m0 = blockIdx.y * (16 * MT);
  int wv = threadIdx.x >> 5, l = threadIdx.x & 31;
  int half = l >> 4, q = l & 15;
  int nb = blockIdx.x * 128 + wv * 16;
  const __bf16* brow = sdb + (size_t)(nb + q) * ISs;
  v8f c[MT] = {};
  for (int k0 = 0; k0 < ISs; k0 += 32) {
    v16bf b = load_b_frag(brow, k0, half);
#pragma unroll
    for (int i = 0; i < MT; ++i) {
      v16bf a = load_a_frag(acts + (size_t)(m0 + 16 * i + q) * ISs, k0, half);
      c[i] = wmma_bf16(a, b, c[i]);
    }
  }
#pragma unroll
  for (int i = 0; i < MT; ++i)
#pragma unroll
    for (int j = 0; j < 8; ++j) {
      int r = m0 + 16 * i + j + 8 * half;
      out[(size_t)r * Hh + nb + q] = c[i][j];
    }
}

// ---------------------------------------------------------------- combine
__global__ void combine_kernel(float* __restrict__ out,
                               const float* __restrict__ ybuf,
                               const float* __restrict__ topw) {
  int idx = blockIdx.x * 256 + threadIdx.x;   // over Ss*Hh
  int t = idx / Hh;
  out[idx] += topw[t * 2 + 0] * ybuf[((size_t)t * 2 + 0) * Hh + (idx % Hh)] +
              topw[t * 2 + 1] * ybuf[((size_t)t * 2 + 1) * Hh + (idx % Hh)];
}

// ---------------------------------------------------------------------------
extern "C" void kernel_launch(void* const* d_in, const int* in_sizes, int n_in,
                              void* d_out, int out_size, void* d_ws, size_t ws_size,
                              hipStream_t stream) {
  const float* x  = (const float*)d_in[0];
  const float* rw = (const float*)d_in[1];
  const float* w1 = (const float*)d_in[2];
  const float* w2 = (const float*)d_in[3];
  const float* sg = (const float*)d_in[4];
  const float* su = (const float*)d_in[5];
  const float* sd = (const float*)d_in[6];
  float* out = (float*)d_out;

  char* ws = (char*)d_ws;
  size_t off = 0;
  auto alloc = [&](size_t bytes) -> void* {
    void* p = ws + off;
    off += (bytes + 255) & ~(size_t)255;
    return p;
  };
  __bf16* xb   = (__bf16*)alloc((size_t)Ss * Hh * 2);
  __bf16* w1b  = (__bf16*)alloc((size_t)Ee * I2 * Hh * 2);
  __bf16* w2b  = (__bf16*)alloc((size_t)Ee * Hh * Ii * 2);
  __bf16* sgb  = (__bf16*)alloc((size_t)ISs * Hh * 2);
  __bf16* sub  = (__bf16*)alloc((size_t)ISs * Hh * 2);
  __bf16* sdb  = (__bf16*)alloc((size_t)Hh * ISs * 2);
  __bf16* actb = (__bf16*)alloc((size_t)2 * Ss * Ii * 2);   // packed 4096 rows
  __bf16* acts = (__bf16*)alloc((size_t)Ss * ISs * 2);
  float*  ybuf = (float*)alloc((size_t)Ss * 2 * Hh * 4);
  float*  topw = (float*)alloc((size_t)Ss * 2 * 4);
  int*    topi = (int*)alloc((size_t)Ss * 2 * 4);
  int*    tok  = (int*)alloc((size_t)Ee * Ss * 4);
  int*    slot = (int*)alloc((size_t)Ee * Ss * 4);
  int*    cnt  = (int*)alloc(256);
  int*    base = (int*)alloc(256);

  // fp32 -> bf16 streaming conversions (bandwidth-bound, ~0.75 GB total).
  auto cvt = [&](const float* s, __bf16* d, int n) {
    cvt_f32_bf16<<<(n / 4 + 255) / 256, 256, 0, stream>>>(s, d, n);
  };
  cvt(x,  xb,  Ss * Hh);
  cvt(w1, w1b, Ee * I2 * Hh);
  cvt(w2, w2b, Ee * Hh * Ii);
  cvt(sg, sgb, ISs * Hh);
  cvt(su, sub, ISs * Hh);
  cvt(sd, sdb, Hh * ISs);

  router_kernel<<<Ss, 256, 0, stream>>>(x, rw, topw, topi);
  build_lists<<<1, 32, 0, stream>>>(topi, tok, slot, cnt, base);

  gemm_gateup_expert<<<dim3(Ii / 128, Ss / (16 * MT), Ee), 256, 0, stream>>>(
      xb, w1b, tok, cnt, base, actb);
  gemm_down_expert<<<dim3(Hh / 128, Ss / (16 * MT), Ee), 256, 0, stream>>>(
      actb, w2b, tok, slot, cnt, base, ybuf);

  gemm_shared_gateup<<<dim3(ISs / 128, Ss / (16 * MT), 1), 256, 0, stream>>>(
      xb, sgb, sub, acts);
  gemm_shared_down<<<dim3(Hh / 128, Ss / (16 * MT), 1), 256, 0, stream>>>(
      acts, sdb, out);

  combine_kernel<<<(Ss * Hh) / 256, 256, 0, stream>>>(out, ybuf, topw);
}
